// Net_35321811042735
// MI455X (gfx1250) — matrix operand order repaired and checked by reading, not compiled, not run on hardware
//
#include <hip/hip_runtime.h>
#include <cstdint>
#include <cstddef>

// ---------------------------------------------------------------------------
// MI455X (gfx1250) fused embedder + autoregressive LSTM decoder.
//
// Roofline: ~146 GFLOP vs ~150 MB of HBM traffic -> compute bound, dominated
// by the 30 recurrent 16x256x96 GEMMs.  Strategy: f16 WMMA
// (v_wmma_f32_16x16x32_f16) with f32 accumulation; one wave32 per 16 batch
// rows; weights packed once into LDS in the ISA WMMA B-operand layout; the
// one-hot token feedback is folded into the WMMA K dimension.  Activations
// use the CDNA5 hardware transcendental V_TANH_F32 so the per-step VALU tail
// does not swamp the 50 WMMAs it has to overlap with.
// ---------------------------------------------------------------------------

typedef _Float16 v16h __attribute__((ext_vector_type(16)));
typedef float    v8f  __attribute__((ext_vector_type(8)));

#define HIDDEN 64
#define NTOK   11
#define MAXLEN 30
#define WAVES  4
#define TPB    (WAVES * 32)
#define ROWS_PER_WG (WAVES * 16)

union AU { uint32_t u[8]; v16h h; };
union H2 { _Float16 h[2]; uint32_t u; };

// ---- fast activations: prefer v_tanh_f32, else v_exp/v_rcp, else libm -----
__device__ inline float fast_tanh(float x) {
#if __has_builtin(__builtin_amdgcn_tanhf)
  return __builtin_amdgcn_tanhf(x);
#elif __has_builtin(__builtin_amdgcn_tanh_f32)
  return __builtin_amdgcn_tanh_f32(x);
#elif __has_builtin(__builtin_amdgcn_exp2f) && __has_builtin(__builtin_amdgcn_rcpf)
  float e = __builtin_amdgcn_exp2f(x * -2.885390082f);   // exp(-2x)
  return 2.0f * __builtin_amdgcn_rcpf(1.0f + e) - 1.0f;
#else
  return tanhf(x);
#endif
}
__device__ inline float sigm(float x) {
#if __has_builtin(__builtin_amdgcn_tanhf) || __has_builtin(__builtin_amdgcn_tanh_f32)
  return 0.5f * fast_tanh(0.5f * x) + 0.5f;
#elif __has_builtin(__builtin_amdgcn_exp2f) && __has_builtin(__builtin_amdgcn_rcpf)
  float e = __builtin_amdgcn_exp2f(x * -1.442695041f);   // exp(-x)
  return __builtin_amdgcn_rcpf(1.0f + e);
#else
  return 0.5f * tanhf(0.5f * x) + 0.5f;
#endif
}

// A-matrix (16x32 f16) element mapping per CDNA5 ISA 7.12.2:
// lanes 0-15: V0..3 -> K 0..7, V4..7 -> K 16..23 ; lanes 16-31: +8
__device__ inline int kAmap(int v, int p, int hi) {
  return (v & 3) * 2 + p + ((v >> 2) << 4) + hi * 8;
}

__device__ inline v8f wmma32(const AU& a, const AU& b, v8f c) {
  return __builtin_amdgcn_wmma_f32_16x16x32_f16(false, a.h, false, b.h,
                                                (short)0, c, false, false);
}

// ---- B-chunk packing: B[k][n] = W[n][k], 32x16 f16 per chunk --------------
// B layout (matches A table order): lane = n, V0..3 -> K 0..7, V4..7 -> K 16..23; lanes 16-31: +8.

// Unchecked fast path: caller guarantees k0+31 < row length and 16 rows valid.
__device__ inline void packB_nc(uint32_t* dst, const float* __restrict__ W,
                                int ldw, int n0, int k0, int lane) {
  const int hi = lane >> 4;
  const int n  = n0 + (lane & 15);
  const float* Wr = W + (size_t)n * ldw + k0 + 8 * hi;
#pragma unroll
  for (int v = 0; v < 8; ++v) {
    H2 t;
    t.h[0] = (_Float16)Wr[(v & 3) * 2 + ((v >> 2) << 4)];
    t.h[1] = (_Float16)Wr[(v & 3) * 2 + ((v >> 2) << 4) + 1];
    dst[lane * 8 + v] = t.u;
  }
}

// Checked path: clamp indices (loads always in-bounds) and select-zero OOB.
__device__ inline void packB_ck(uint32_t* dst, const float* __restrict__ W,
                                int ldw, int n0, int k0, int kmax, int nmax,
                                int lane) {
  const int hi = lane >> 4;
  const int nn = n0 + (lane & 15);
  const bool nok = nn < nmax;
  const float* Wr = W + (size_t)(nok ? nn : 0) * ldw;
#pragma unroll
  for (int v = 0; v < 8; ++v) {
    H2 t;
#pragma unroll
    for (int p = 0; p < 2; ++p) {
      int k = k0 + kAmap(v, p, hi);
      float f = Wr[k < kmax ? k : 0];          // unconditional clamped load
      t.h[p] = (_Float16)((k < kmax && nok) ? f : 0.0f);
    }
    dst[lane * 8 + v] = t.u;
  }
}

__device__ inline AU loadB(const uint32_t* src, int lane) {
  AU b;
  const uint32_t* p = src + lane * 8;
#pragma unroll
  for (int v = 0; v < 8; ++v) b.u[v] = p[v];   // -> 2x ds_load_b128
  return b;
}

// Build an A chunk (16x32 f16) for this wave from an LDS [row][k] f16 tile.
__device__ inline AU loadA_h(const _Float16* hrow, int lane, int kc) {
  AU a;
  int m = lane & 15, hi = lane >> 4;
#pragma unroll
  for (int v = 0; v < 8; ++v) {
    int k = kc * 32 + (v & 3) * 2 + ((v >> 2) << 4) + hi * 8;  // even -> pair
    a.u[v] = *(const uint32_t*)&hrow[m * HIDDEN + k];          // 2x b128
  }
  return a;
}

// Pack two f32 quads (8 consecutive K values) into A VGPRs v0..v3 / v4..v7.
__device__ inline void packA_quads(AU& a, int vbase, float4 qa, float4 qb) {
  H2 t;
  t.h[0] = (_Float16)qa.x; t.h[1] = (_Float16)qa.y; a.u[vbase + 0] = t.u;
  t.h[0] = (_Float16)qa.z; t.h[1] = (_Float16)qa.w; a.u[vbase + 1] = t.u;
  t.h[0] = (_Float16)qb.x; t.h[1] = (_Float16)qb.y; a.u[vbase + 2] = t.u;
  t.h[0] = (_Float16)qb.z; t.h[1] = (_Float16)qb.w; a.u[vbase + 3] = t.u;
}

__global__ __launch_bounds__(TPB, 1) void net_lstm_kernel(
    const float* __restrict__ x,   const float* __restrict__ w1,
    const float* __restrict__ b1,  const float* __restrict__ w2,
    const float* __restrict__ b2,  const float* __restrict__ w3,
    const float* __restrict__ b3,  const float* __restrict__ w_ih,
    const float* __restrict__ w_hh,const float* __restrict__ b_ih,
    const float* __restrict__ b_hh,const float* __restrict__ w_out,
    const float* __restrict__ b_out, float* __restrict__ out, int Bsz) {
  // --- LDS: 49152 + 4096 + 2048 + 8192 + 256 + 1024 = 64768 B ---
  __shared__ uint32_t recW[48 * 256];   // 16 gate tiles x 3 K-chunks (K=96)
  __shared__ uint32_t embW[4 * 256];    // rotating embedder B buffer (4 tiles)
  __shared__ uint32_t outWs[2 * 256];   // head: 1 N-tile x 2 K-chunks
  __shared__ _Float16 hS[WAVES * 16 * HIDDEN];  // per-wave h handoff (A src)
  __shared__ int      topS[WAVES * 16];         // per-row argmax token
  __shared__ float    biasg[256];               // b_ih + b_hh

  const int tid  = threadIdx.x;
  const int lane = tid & 31;
  const int wave = tid >> 5;
  const int grp  = wave;           // packing group
  const int m    = lane & 15;
  const int hi   = lane >> 4;
  const int rowBase = blockIdx.x * ROWS_PER_WG + wave * 16;
  _Float16* hw = &hS[wave * 16 * HIDDEN];
  const v8f czero = {};            // WMMA C = inline 0

  // ================= Embedder layer 1: relu(x @ w1^T + b1), K=220 =========
  v8f eacc[4];
#pragma unroll
  for (int nt = 0; nt < 4; ++nt) eacc[nt] = czero;

  const float* xrow = x + (size_t)(rowBase + m) * 220;  // 16B aligned (880B)
  const float* xr   = xrow + hi * 8;

#pragma unroll 1
  for (int kc = 0; kc < 6; ++kc) {   // chunks 0..5: fully in-range
    packB_nc(&embW[grp * 256], w1, 220, grp * 16, kc * 32, lane);
    __syncthreads();
    AU a;
    packA_quads(a, 0, *(const float4*)(xr + kc * 32 + 0),
                      *(const float4*)(xr + kc * 32 + 4));
    packA_quads(a, 4, *(const float4*)(xr + kc * 32 + 16),
                      *(const float4*)(xr + kc * 32 + 20));
#pragma unroll
    for (int nt = 0; nt < 4; ++nt)
      eacc[nt] = wmma32(a, loadB(&embW[nt * 256], lane), eacc[nt]);
    __syncthreads();
  }
  {  // tail chunk kc==6: K 192..219 valid; only hi==1,v>=6 (K 220..223) OOB
    packB_ck(&embW[grp * 256], w1, 220, grp * 16, 192, 220, 64, lane);
    __syncthreads();
    AU a;
    packA_quads(a, 0, *(const float4*)(xr + 192), *(const float4*)(xr + 196));
    float4 q2 = *(const float4*)(xr + 208);           // hi=1 -> K 216..219 ok
    float4 q3 = *(const float4*)(xrow + 212);         // always in-bounds
    if (hi) { q3.x = 0.f; q3.y = 0.f; q3.z = 0.f; q3.w = 0.f; }
    packA_quads(a, 4, q2, q3);
#pragma unroll
    for (int nt = 0; nt < 4; ++nt)
      eacc[nt] = wmma32(a, loadB(&embW[nt * 256], lane), eacc[nt]);
    __syncthreads();
  }
#pragma unroll
  for (int nt = 0; nt < 4; ++nt) {
    float bn = b1[nt * 16 + m];
#pragma unroll
    for (int r = 0; r < 8; ++r) {
      float e = eacc[nt][r] + bn;
      hw[(r + 8 * hi) * HIDDEN + nt * 16 + m] = (_Float16)(e > 0.f ? e : 0.f);
    }
  }
  __syncthreads();

  // ================= Embedder layers 2 & 3: relu(h @ W^T + b), K=64 =======
#pragma unroll
  for (int layer = 0; layer < 2; ++layer) {
    const float* W  = (layer == 0) ? w2 : w3;
    const float* bv = (layer == 0) ? b2 : b3;
    AU a0 = loadA_h(hw, lane, 0);     // buffer A in regs before hS overwrite
    AU a1 = loadA_h(hw, lane, 1);
#pragma unroll
    for (int nt = 0; nt < 4; ++nt) eacc[nt] = czero;
#pragma unroll 1
    for (int kc = 0; kc < 2; ++kc) {
      packB_nc(&embW[grp * 256], W, 64, grp * 16, kc * 32, lane);
      __syncthreads();
      AU a = (kc == 0) ? a0 : a1;
#pragma unroll
      for (int nt = 0; nt < 4; ++nt)
        eacc[nt] = wmma32(a, loadB(&embW[nt * 256], lane), eacc[nt]);
      __syncthreads();
    }
#pragma unroll
    for (int nt = 0; nt < 4; ++nt) {
      float bn = bv[nt * 16 + m];
#pragma unroll
      for (int r = 0; r < 8; ++r) {
        float e = eacc[nt][r] + bn;
        hw[(r + 8 * hi) * HIDDEN + nt * 16 + m] = (_Float16)(e > 0.f ? e : 0.f);
      }
    }
    __syncthreads();
  }

  // ================= Pack LSTM weights =====================================
  // Step 0 uses w_ih over K = [emb(64) | token(11) | pad] -> 3 chunks of 32.
  for (int u = grp; u < 48; u += WAVES) {
    int gt = u / 3, ch = u % 3;
    if (ch < 2) packB_nc(&recW[u * 256], w_ih, 75, gt * 16, ch * 32, lane);
    else        packB_ck(&recW[u * 256], w_ih, 75, gt * 16, 64, 75, 256, lane);
  }
  if (grp < 2)  // head: logits = h @ w_out^T, rows >= 11 zero-padded
    packB_ck(&outWs[grp * 256], w_out, 64, 0, grp * 32, 64, NTOK, lane);
  for (int i = tid; i < 256; i += TPB) biasg[i] = b_ih[i] + b_hh[i];
  __syncthreads();

  float biasv[16];
#pragma unroll
  for (int gt = 0; gt < 16; ++gt) biasv[gt] = biasg[gt * 16 + m];
  const float boutv = (m < NTOK) ? b_out[m] : -1e30f;  // pad cols never argmax

  v8f cst[4];
#pragma unroll
  for (int i = 0; i < 4; ++i) cst[i] = czero;

  // Incrementally-advanced per-lane output pointers (immediate row offsets).
  const size_t progBase = (size_t)MAXLEN * Bsz * NTOK;
  float* pProb = out + (size_t)(rowBase + 8 * hi) * NTOK + m;
  float* pProg = out + progBase + rowBase + 8 * hi;

  // a0/a1 carry the A-layout activation operand across phases:
  // entering step t they hold h_{t-1} (or the embedding e at t==0).
  AU a0 = loadA_h(hw, lane, 0);
  AU a1 = loadA_h(hw, lane, 1);

  // One LSTM step: gate WMMA chains, activations, head, outputs, argmax.
  auto lstmStep = [&](bool useTok) {
    AU a2;                         // one-hot token chunk (steps >= 1 only)
#pragma unroll
    for (int v = 0; v < 8; ++v) a2.u[v] = 0u;
    if (useTok) {
      int tok = topS[wave * 16 + m];
#pragma unroll
      for (int v = 0; v < 8; ++v) {
        H2 tt;
#pragma unroll
        for (int p = 0; p < 2; ++p)
          tt.h[p] = (kAmap(v, p, hi) == tok) ? (_Float16)1.0f : (_Float16)0.0f;
        a2.u[v] = tt.u;
      }
    }

    // gates (16x256) in 4 hidden-tile groups: i|f|g|o tiles ht,4+ht,8+ht,12+ht
#pragma unroll
    for (int ht = 0; ht < 4; ++ht) {
      v8f gacc[4];
#pragma unroll
      for (int gs = 0; gs < 4; ++gs) {
        int gt = gs * 4 + ht;
        v8f g = wmma32(a0, loadB(&recW[(gt * 3 + 0) * 256], lane), czero);
        g = wmma32(a1, loadB(&recW[(gt * 3 + 1) * 256], lane), g);
        if (useTok)
          g = wmma32(a2, loadB(&recW[(gt * 3 + 2) * 256], lane), g);
        gacc[gs] = g;
      }
#pragma unroll
      for (int r = 0; r < 8; ++r) {   // bias folded into activation input
        float iv = sigm(gacc[0][r] + biasv[ht]);
        float fv = sigm(gacc[1][r] + biasv[4 + ht]);
        float gv = fast_tanh(gacc[2][r] + biasv[8 + ht]);
        float ov = sigm(gacc[3][r] + biasv[12 + ht]);
        float c  = fv * cst[ht][r] + iv * gv;
        cst[ht][r] = c;
        hw[(r + 8 * hi) * HIDDEN + ht * 16 + m] =
            (_Float16)(ov * fast_tanh(c));
      }
    }

    // reload h_t in A layout; these registers feed BOTH the head GEMM below
    // and the next step's gate GEMMs (no reload at the top of the next step).
    a0 = loadA_h(hw, lane, 0);
    a1 = loadA_h(hw, lane, 1);

    v8f lg = wmma32(a0, loadB(&outWs[0], lane), czero);
    lg = wmma32(a1, loadB(&outWs[256], lane), lg);

#pragma unroll
    for (int r = 0; r < 8; ++r) {
      float logit = lg[r] + boutv;
      if (m < NTOK) pProb[r * NTOK] = logit;
      // in-wave argmax over the 16-lane group (first-index tie-break)
      float val = logit;
      int   idx = m;
#pragma unroll
      for (int off = 8; off >= 1; off >>= 1) {
        float ov = __shfl_xor(val, off, 32);
        int   oi = __shfl_xor(idx, off, 32);
        bool take = (ov > val) || (ov == val && oi < idx);
        val = take ? ov : val;
        idx = take ? oi : idx;
      }
      if (m == 0) {
        topS[wave * 16 + r + 8 * hi] = idx;
        // program tokens emitted as float (d_out is a single float buffer)
        pProg[r] = (float)idx;
      }
    }
    pProb += (size_t)Bsz * NTOK;
    pProg += Bsz;
  };

  // ---- step 0 (token chunk is all-zero: 2-WMMA chains) --------------------
  lstmStep(false);
  __syncthreads();
  // swap recurrent K-chunks 0/1 from w_ih[:, :64] to w_hh for steps >= 1
  for (int u = grp; u < 32; u += WAVES) {
    int gt = u >> 1, ch = u & 1;
    packB_nc(&recW[(gt * 3 + ch) * 256], w_hh, 64, gt * 16, ch * 32, lane);
  }
  __syncthreads();

  // ---- steps 1..29: steady-state hot loop ---------------------------------
#pragma unroll 1
  for (int t = 1; t < MAXLEN; ++t) {
    lstmStep(true);
    __syncthreads();
  }
}

extern "C" void kernel_launch(void* const* d_in, const int* in_sizes, int n_in,
                              void* d_out, int out_size, void* d_ws,
                              size_t ws_size, hipStream_t stream) {
  const float* x     = (const float*)d_in[0];
  const float* w1    = (const float*)d_in[1];
  const float* b1    = (const float*)d_in[2];
  const float* w2    = (const float*)d_in[3];
  const float* b2    = (const float*)d_in[4];
  const float* w3    = (const float*)d_in[5];
  const float* b3    = (const float*)d_in[6];
  const float* w_ih  = (const float*)d_in[7];
  const float* w_hh  = (const float*)d_in[8];
  const float* b_ih  = (const float*)d_in[9];
  const float* b_hh  = (const float*)d_in[10];
  const float* w_out = (const float*)d_in[11];
  const float* b_out = (const float*)d_in[12];

  int Bsz    = in_sizes[0] / 220;      // 65536
  int blocks = Bsz / ROWS_PER_WG;      // 1024
  net_lstm_kernel<<<blocks, TPB, 0, stream>>>(x, w1, b1, w2, b2, w3, b3, w_ih,
                                              w_hh, b_ih, b_hh, w_out, b_out,
                                              (float*)d_out, Bsz);
}
